// HausdorffDTLoss_16604343566861
// MI455X (gfx1250) — matrix operand order repaired and checked by reading, not compiled, not run on hardware
//
#include <hip/hip_runtime.h>
#include <hip/hip_bf16.h>
#include <math.h>

// Problem constants (from reference setup_inputs)
#define R_   128
#define C_   80
#define H_   28
#define W_   28
#define P_   784          // H*W
#define PP_  800          // P padded to multiple of 32 (WMMA K-step)
#define INF_ 1e30f

typedef _Float16 v16h __attribute__((ext_vector_type(16)));
typedef _Float16 v8h  __attribute__((ext_vector_type(8)));
typedef float    v8f  __attribute__((ext_vector_type(8)));

// ---------------------------------------------------------------------------
// Kernel A: one workgroup per ROI. Fused: sigmoid+gather, exact separable EDT
// (two 1-D min-plus passes == brute-force min over d2), field normalization,
// per-pixel dist/err, converted to f16 rows [R, PP_] for the WMMA contraction.
// ---------------------------------------------------------------------------
__global__ __launch_bounds__(256)
void hdt_field_kernel(const float* __restrict__ preds,
                      const float* __restrict__ targets,
                      const int*   __restrict__ labels,
                      _Float16* __restrict__ errOut,
                      _Float16* __restrict__ distOut)
{
    __shared__ float sImg[2][P_];     // [0]=sigmoid(pred_sel), [1]=target
    __shared__ float sG[P_];          // pass-1 result (per-column min)
    __shared__ float sD[P_];          // pass-2 result / edt
    __shared__ float sField[2][P_];   // accumulated distance fields
    __shared__ float sRed[256];
    __shared__ float sBcast;

    const int tid = threadIdx.x;
    const int r   = blockIdx.x;
    const int lab = labels[r];
    const float* pp = preds   + ((size_t)r * C_ + (size_t)lab) * P_;
    const float* tt = targets + (size_t)r * P_;

    for (int p = tid; p < P_; p += 256) {
        float x = pp[p];
        sImg[0][p] = 1.0f / (1.0f + expf(-x));
        sImg[1][p] = tt[p];
        sField[0][p] = 0.0f;
        sField[1][p] = 0.0f;
    }
    __syncthreads();

    for (int img = 0; img < 2; ++img) {
        // count foreground pixels -> fg_any / bg_any flags
        float cnt = 0.0f;
        for (int p = tid; p < P_; p += 256)
            cnt += (sImg[img][p] > 0.5f) ? 1.0f : 0.0f;
        sRed[tid] = cnt;
        __syncthreads();
        if (tid == 0) {
            float s = 0.0f;
            for (int i = 0; i < 256; ++i) s += sRed[i];
            sBcast = s;
        }
        __syncthreads();
        const bool fgAny = (sBcast > 0.5f);
        const bool bgAny = (sBcast < (float)P_ - 0.5f);

        for (int ms = 0; ms < 2; ++ms) {   // ms=0: edt(fg), ms=1: edt(~fg)
            // Pass 1 (vertical): sG[y,x] = min_{y'} (y-y')^2 + penalty(y',x)
            for (int p = tid; p < P_; p += 256) {
                int y = p / W_, x = p - y * W_;
                float best = INF_;
                #pragma unroll
                for (int yp = 0; yp < H_; ++yp) {
                    bool fg = sImg[img][yp * W_ + x] > 0.5f;
                    bool masked = (ms == 0) ? fg : !fg;
                    float dy = (float)(y - yp);
                    float v = dy * dy;
                    if (!masked && v < best) best = v;
                }
                sG[p] = best;
            }
            __syncthreads();
            // Pass 2 (horizontal): sD[y,x] = min_{x'} (x-x')^2 + sG[y,x']
            for (int p = tid; p < P_; p += 256) {
                int y = p / W_, x = p - y * W_;
                float best = INF_;
                const float* grow = &sG[y * W_];
                #pragma unroll
                for (int xp = 0; xp < W_; ++xp) {
                    float dx = (float)(x - xp);
                    float v = dx * dx + grow[xp];
                    if (v < best) best = v;
                }
                // any(~mask) false <=> min stayed ~INF (true d2 <= 1458)
                sD[p] = (best > 1e20f) ? 0.0f : sqrtf(best);
            }
            __syncthreads();
            // max-reduce of edt, clamp EPS
            float mx = 0.0f;
            for (int p = tid; p < P_; p += 256) mx = fmaxf(mx, sD[p]);
            sRed[tid] = mx;
            __syncthreads();
            if (tid == 0) {
                float m = 0.0f;
                for (int i = 0; i < 256; ++i) m = fmaxf(m, sRed[i]);
                sBcast = fmaxf(m, 1e-12f);
            }
            __syncthreads();
            const float mmax = sBcast;
            const bool cond = (ms == 0) ? fgAny : (fgAny && bgAny);
            const float scale = cond ? (1.0f / mmax) : 0.0f;
            for (int p = tid; p < P_; p += 256)
                sField[img][p] += sD[p] * scale;
            __syncthreads();
        }
    }

    // dist = field_p^2 + field_t^2 (ALPHA=2), err = (p - t)^2 ; emit f16 rows
    _Float16* eo = errOut  + (size_t)r * PP_;
    _Float16* dd = distOut + (size_t)r * PP_;
    for (int p = tid; p < P_; p += 256) {
        float fp = sField[0][p], ft = sField[1][p];
        float dv = fp * fp + ft * ft;
        float ev = sImg[0][p] - sImg[1][p];
        ev = ev * ev;
        eo[p] = (_Float16)ev;
        dd[p] = (_Float16)dv;
    }
    for (int p = P_ + tid; p < PP_; p += 256) {   // zero K padding
        eo[p] = (_Float16)0.0f;
        dd[p] = (_Float16)0.0f;
    }
}

// ---------------------------------------------------------------------------
// Kernel B: triangular contraction  S = sum_{r' <= r} sum_p err[r,p]*dist[r',p]
// via v_wmma_f32_16x16x32_f16. 8x8 grid of 16x16 tiles; only I>=J computed
// (36 tiles), diagonal tiles masked per-lane with the documented C/D layout.
// One block, 8 wave32s, deterministic LDS reduction, single scalar store.
// ---------------------------------------------------------------------------
__global__ __launch_bounds__(256)
void hdt_wmma_reduce_kernel(const _Float16* __restrict__ err,
                            const _Float16* __restrict__ dist,
                            float* __restrict__ out)
{
    __shared__ float sPart[256];
    const int tid    = threadIdx.x;
    const int wave   = tid >> 5;
    const int lane   = tid & 31;
    const int laneHi = (lane >> 4) & 1;   // 0: lanes 0-15, 1: lanes 16-31
    const int laneLo = lane & 15;

    float local = 0.0f;
    int t = 0;
    for (int I = 0; I < 8; ++I) {
        for (int J = 0; J <= I; ++J, ++t) {
            if ((t & 7) != wave) continue;

            // A (err, 16x32 f16): lane holds row M=laneLo; two 16B chunks at
            //   K = k + 8*laneHi (+0) and K = k + 16 + 8*laneHi.
            // B (dist, 32x16 f16): lane holds col N=laneLo; two 16B chunks at
            //   K = k + 8*laneHi (+0) and K = k + 16 + 8*laneHi.
            const _Float16* aBase = err  + (size_t)(I * 16 + laneLo) * PP_ + laneHi * 8;
            const _Float16* bBase = dist + (size_t)(J * 16 + laneLo) * PP_ + laneHi * 8;

            v8f c = {};
            for (int k = 0; k < PP_; k += 32) {
                v8h a0 = *(const v8h*)(aBase + k);
                v8h a1 = *(const v8h*)(aBase + k + 16);
                v8h b0 = *(const v8h*)(bBase + k);
                v8h b1 = *(const v8h*)(bBase + k + 16);
                v16h a = __builtin_shufflevector(a0, a1,
                          0,1,2,3,4,5,6,7,8,9,10,11,12,13,14,15);
                v16h b = __builtin_shufflevector(b0, b1,
                          0,1,2,3,4,5,6,7,8,9,10,11,12,13,14,15);
                c = __builtin_amdgcn_wmma_f32_16x16x32_f16(
                        /*neg_a=*/false, a, /*neg_b=*/false, b,
                        /*c_mod=*/(short)0, c,
                        /*reuse_a=*/false, /*reuse_b=*/false);
            }

            // C/D layout: VGPR d -> M = d + 8*laneHi, N = laneLo.
            // Keep element iff global r (M-dim, err) >= global r' (N-dim, dist).
            const int Ng = J * 16 + laneLo;
            #pragma unroll
            for (int d = 0; d < 8; ++d) {
                int Mg = I * 16 + d + laneHi * 8;
                if (Mg >= Ng) local += c[d];
            }
        }
    }

    sPart[tid] = local;
    __syncthreads();
    if (tid == 0) {
        float s = 0.0f;
        for (int i = 0; i < 256; ++i) s += sPart[i];
        out[0] = s * (1.0f / ((float)R_ * (float)P_));
    }
}

// ---------------------------------------------------------------------------
extern "C" void kernel_launch(void* const* d_in, const int* in_sizes, int n_in,
                              void* d_out, int out_size, void* d_ws, size_t ws_size,
                              hipStream_t stream)
{
    const float* preds   = (const float*)d_in[0];   // [128,80,28,28] f32
    const float* targets = (const float*)d_in[1];   // [128,28,28]    f32
    const int*   labels  = (const int*)d_in[2];     // [128]          i32
    float*       out     = (float*)d_out;           // scalar f32

    _Float16* errW  = (_Float16*)d_ws;                         // [128, 800] f16
    _Float16* distW = errW + (size_t)R_ * PP_;                 // [128, 800] f16
    // total workspace used: 2 * 128 * 800 * 2 = 409,600 bytes

    hdt_field_kernel<<<R_, 256, 0, stream>>>(preds, targets, labels, errW, distW);
    hdt_wmma_reduce_kernel<<<1, 256, 0, stream>>>(errW, distW, out);
}
